// RandSelfAttention2D_80333068305049
// MI455X (gfx1250) — matrix operand order repaired and checked by reading, not compiled, not run on hardware
//
#include <hip/hip_runtime.h>

// ---------------- constants ----------------
#define BB   4
#define CC   64
#define NN   16384      // 128*128
#define MM   1638       // floor(0.1 * N)
#define MPAD 1664       // 52 * 32
#define DD   64
#define KSTR 72         // padded LDS row stride (bf16 elems) for K panel
#define VSTR 40         // padded LDS row stride (bf16 elems) for V tile

typedef __bf16 bf16;
typedef __attribute__((ext_vector_type(8)))  bf16  v8bf;
typedef __attribute__((ext_vector_type(16))) bf16  v16bf;
typedef __attribute__((ext_vector_type(8)))  float v8f;

__device__ __forceinline__ v8f wmma_bf16(v16bf a, v16bf b, v8f c) {
  // v_wmma_f32_16x16x32_bf16 : D = A(16x32) * B(32x16) + C(16x16 f32)
  return __builtin_amdgcn_wmma_f32_16x16x32_bf16(
      /*neg_a=*/false, a, /*neg_b=*/false, b,
      /*c_mod=*/(short)0, c, /*reuse_a=*/false, /*reuse_b=*/false);
}

__device__ __forceinline__ v16bf join16(v8bf lo, v8bf hi) {
  return __builtin_shufflevector(lo, hi, 0,1,2,3,4,5,6,7,8,9,10,11,12,13,14,15);
}

// A operand (16x32 bf16) from LDS tile laid out [row][64]:
//   lanes 0-15 : row = base + lane,      K = ks*32 + 0..15
//   lanes16-31 : row = base + lane-16,   K = ks*32 + 16..31
__device__ __forceinline__ v16bf load_A_lds(const bf16* ldsx, int rbase, int lane, int ks) {
  const bf16* r = ldsx + (size_t)(rbase + (lane & 15)) * 64 + ks * 32 + ((lane >> 4) * 16);
  return join16(*(const v8bf*)(r), *(const v8bf*)(r + 8));
}

// B operand (32x16 bf16) from LDS weight stored transposed: wt[col][64]:
//   lanes 0-15 : col = jt*16+lane,     K = ks*32 + 0..15
//   lanes16-31 : col = jt*16+lane-16,  K = ks*32 + 16..31
__device__ __forceinline__ v16bf load_B_lds(const bf16* wt, int jt, int lane, int ks) {
  const bf16* r = wt + (size_t)(jt * 16 + (lane & 15)) * 64 + ks * 32 + ((lane >> 4) * 16);
  return join16(*(const v8bf*)(r), *(const v8bf*)(r + 8));
}

// ---------------- kernel: zero key-mass accumulator ----------------
__global__ void k_zero_ssum(float* ssum) {
  int i = blockIdx.x * 256 + threadIdx.x;
  if (i < BB * MPAD) ssum[i] = 0.0f;
}

// ---------------- kernel: Q projection (all N rows) ----------------
// grid = B * N/128 = 512 blocks, 256 threads (8 waves, 16 rows each)
__global__ __launch_bounds__(256) void k_proj_q(const float* __restrict__ x,
                                                const float* __restrict__ Wq,
                                                bf16* __restrict__ q_bf) {
  __shared__ __align__(32) bf16 s_wt[64 * 64];   // Wq transposed: [j][c]
  __shared__ __align__(32) bf16 s_x[128 * 64];   // xf tile: [n_local][c]
  int b  = blockIdx.x >> 7;
  int n0 = (blockIdx.x & 127) * 128;
  int t  = threadIdx.x;

  for (int e = t; e < 4096; e += 256) {
    int c = e >> 6, j = e & 63;
    s_wt[j * 64 + c] = (bf16)Wq[e];
  }
  const float* xb = x + (size_t)b * CC * NN;
  for (int e = t; e < 8192; e += 256) {
    int c = e >> 7, nl = e & 127;                 // coalesced over nl
    s_x[nl * 64 + c] = (bf16)xb[(size_t)c * NN + n0 + nl];
  }
  __syncthreads();

  int wave = t >> 5, lane = t & 31;
  int rbase = wave * 16;
  v16bf a0 = load_A_lds(s_x, rbase, lane, 0);
  v16bf a1 = load_A_lds(s_x, rbase, lane, 1);
  int r0 = rbase + ((lane >> 4) << 3);
  #pragma unroll
  for (int jt = 0; jt < 4; jt++) {
    v8f acc = {};
    acc = wmma_bf16(a0, load_B_lds(s_wt, jt, lane, 0), acc);
    acc = wmma_bf16(a1, load_B_lds(s_wt, jt, lane, 1), acc);
    int j = jt * 16 + (lane & 15);
    bf16* qp = q_bf + ((size_t)b * NN + n0 + r0) * 64 + j;
    #pragma unroll
    for (int v = 0; v < 8; v++) qp[(size_t)v * 64] = (bf16)acc[v];
  }
}

// ---------------- kernel: K/V projection at gathered rows only ----------------
// grid = B * MPAD/128 = 52 blocks, 256 threads
__global__ __launch_bounds__(256) void k_proj_kv(const float* __restrict__ x,
                                                 const float* __restrict__ Wk,
                                                 const float* __restrict__ Wv,
                                                 const int* __restrict__ idx,
                                                 bf16* __restrict__ k_bf,
                                                 bf16* __restrict__ vt_bf) {
  __shared__ __align__(32) bf16 s_wk[64 * 64];
  __shared__ __align__(32) bf16 s_wv[64 * 64];
  __shared__ __align__(32) bf16 s_x[128 * 64];
  __shared__ int s_idx[128];
  int b  = blockIdx.x / 13;
  int m0 = (blockIdx.x % 13) * 128;
  int t  = threadIdx.x;

  for (int e = t; e < 4096; e += 256) {
    int c = e >> 6, j = e & 63;
    s_wk[j * 64 + c] = (bf16)Wk[e];
    s_wv[j * 64 + c] = (bf16)Wv[e];
  }
  if (t < 128) {
    int m = m0 + t;
    s_idx[t] = (m < MM) ? idx[b * MM + m] : -1;
  }
  __syncthreads();

  const float* xb = x + (size_t)b * CC * NN;
  for (int e = t; e < 8192; e += 256) {
    int ml = e >> 6, c = e & 63;
    int n = s_idx[ml];
    s_x[ml * 64 + c] = (n >= 0) ? (bf16)xb[(size_t)c * NN + n] : (bf16)0.0f;
  }
  __syncthreads();

  int wave = t >> 5, lane = t & 31;
  int rbase = wave * 16;
  v16bf a0 = load_A_lds(s_x, rbase, lane, 0);
  v16bf a1 = load_A_lds(s_x, rbase, lane, 1);
  int r0 = rbase + ((lane >> 4) << 3);
  #pragma unroll
  for (int jt = 0; jt < 4; jt++) {
    int j = jt * 16 + (lane & 15);
    {   // keys, row-major [m][d]
      v8f acc = {};
      acc = wmma_bf16(a0, load_B_lds(s_wk, jt, lane, 0), acc);
      acc = wmma_bf16(a1, load_B_lds(s_wk, jt, lane, 1), acc);
      bf16* kp = k_bf + ((size_t)b * MPAD + m0 + r0) * 64 + j;
      #pragma unroll
      for (int v = 0; v < 8; v++) kp[(size_t)v * 64] = (bf16)acc[v];
    }
    {   // values, transposed [d][m]  (B-operand-friendly for P@V)
      v8f acc = {};
      acc = wmma_bf16(a0, load_B_lds(s_wv, jt, lane, 0), acc);
      acc = wmma_bf16(a1, load_B_lds(s_wv, jt, lane, 1), acc);
      v8bf pk;
      #pragma unroll
      for (int v = 0; v < 8; v++) pk[v] = (bf16)acc[v];
      *(v8bf*)(vt_bf + ((size_t)b * 64 + j) * MPAD + m0 + r0) = pk;
    }
  }
}

// ---------------- kernel: attention ----------------
// 256 blocks * 512 threads; 16 waves/block, one 16-query tile per wave.
// Whole K panel staged once into LDS via async global->LDS; V tiles streamed
// into LDS per 32-key step, overlapped with the score WMMAs.
__global__ __launch_bounds__(512) void k_attn(const bf16* __restrict__ q_bf,
                                              const bf16* __restrict__ k_bf,
                                              const bf16* __restrict__ vt_bf,
                                              float* __restrict__ out,
                                              float* __restrict__ ssum_g) {
  __shared__ __align__(32) bf16  s_k[MPAD * KSTR];   // 239,616 B  K panel (padded rows)
  __shared__ __align__(32) bf16  s_v[64 * VSTR];     //   5,120 B  V tile  (padded rows)
  __shared__ __align__(32) float s_sum[MPAD];        //   6,656 B  per-key mass
  __shared__ __align__(32) bf16  s_p[16 * 512];      //  16,384 B  per-wave P transpose
  int t     = threadIdx.x;
  int b     = blockIdx.x >> 6;
  int n0    = ((blockIdx.x & 63) << 8) + ((t >> 5) << 4);  // this wave's query tile
  int lane  = t & 31;
  int mcol  = lane & 15;
  int khalf = lane >> 4;

  // ---- stage whole K panel into LDS with async global->LDS copies ----
  {
    uint32_t lbase = (uint32_t)(uintptr_t)(&s_k[0]);
    const bf16* gk = k_bf + (size_t)b * MPAD * 64;
    for (int i = t; i < MPAD * 8; i += 512) {        // 16B chunks, 8 per 64-elem row
      int m = i >> 3, c = i & 7;
      uint32_t loff = lbase + (uint32_t)(m * KSTR + c * 8) * 2;
      const bf16* gp = gk + (size_t)m * 64 + c * 8;
      asm volatile("global_load_async_to_lds_b128 %0, %1, off"
                   :: "v"(loff), "v"(gp) : "memory");
    }
  }
  for (int i = t; i < MPAD; i += 512) s_sum[i] = 0.0f;

  // ---- Q tile operand (held in registers for the whole kernel) ----
  const bf16* qrow = q_bf + ((size_t)b * NN + n0 + mcol) * 64;
  int cb = khalf * 16;
  v16bf a0 = join16(*(const v8bf*)(qrow + cb),      *(const v8bf*)(qrow + cb + 8));
  v16bf a1 = join16(*(const v8bf*)(qrow + 32 + cb), *(const v8bf*)(qrow + 32 + cb + 8));

  asm volatile("s_wait_asynccnt 0x0" ::: "memory");
  __syncthreads();                                   // K panel + s_sum ready

  const bf16* kb = s_k + (size_t)mcol * KSTR + khalf * 16;
  const float scale = 0.125f;                        // 1/sqrt(64)

  // ---- pass 1: per-lane online max / sum-exp over this lane's key columns ----
  float lmax[8], lsum[8];
  #pragma unroll
  for (int v = 0; v < 8; v++) { lmax[v] = -1e30f; lsum[v] = 0.0f; }
  for (int mt = 0; mt < MPAD / 16; mt++) {
    const bf16* kr = kb + (size_t)mt * 16 * KSTR;
    v16bf b0 = join16(*(const v8bf*)(kr),      *(const v8bf*)(kr + 8));
    v16bf b1 = join16(*(const v8bf*)(kr + 32), *(const v8bf*)(kr + 40));
    v8f acc = {};
    acc = wmma_bf16(a0, b0, acc);
    acc = wmma_bf16(a1, b1, acc);
    bool valid = (mt * 16 + mcol) < MM;
    #pragma unroll
    for (int v = 0; v < 8; v++) {
      float s  = valid ? acc[v] * scale : -2e30f;    // masked -> exp underflows to 0
      float nm = fmaxf(lmax[v], s);
      lsum[v]  = lsum[v] * __expf(lmax[v] - nm) + __expf(s - nm);
      lmax[v]  = nm;
    }
  }
  // cross-lane merge within 16-lane groups (rows 0-7 in lanes 0-15, rows 8-15 in 16-31)
  float rmax[8], rinv[8];
  #pragma unroll
  for (int v = 0; v < 8; v++) {
    float mx = lmax[v], sm = lsum[v];
    #pragma unroll
    for (int d = 1; d < 16; d <<= 1) {
      float omx = __shfl_xor(mx, d, 16);
      float osm = __shfl_xor(sm, d, 16);
      float nm  = fmaxf(mx, omx);
      sm = sm * __expf(mx - nm) + osm * __expf(omx - nm);
      mx = nm;
    }
    rmax[v] = mx;
    rinv[v] = 1.0f / sm;
  }

  // ---- pass 2 (lockstep over 32-key steps): recompute logits, normalize,
  //      P@V via WMMA from an async-staged V tile, accumulate key mass ----
  v8f oacc[4] = {{}, {}, {}, {}};
  bf16* myp = s_p + (size_t)(t >> 5) * 512;          // this wave's 16x32 P buffer
  int prow0 = khalf * 8;
  uint32_t vbase = (uint32_t)(uintptr_t)(&s_v[0]);
  const bf16* vtb = vt_bf + (size_t)b * 64 * MPAD;
  for (int ms = 0; ms < MPAD / 32; ms++) {
    __syncthreads();                                 // previous V tile fully consumed
    if (t < 256) {                                   // async-stage V tile for this step
      int d = t >> 2, c = t & 3;
      uint32_t loff = vbase + (uint32_t)(d * VSTR + c * 8) * 2;
      const bf16* gp = vtb + (size_t)d * MPAD + ms * 32 + c * 8;
      asm volatile("global_load_async_to_lds_b128 %0, %1, off"
                   :: "v"(loff), "v"(gp) : "memory");
    }
    // score WMMAs overlap with the in-flight V tile
    #pragma unroll
    for (int h = 0; h < 2; h++) {
      int mt = ms * 2 + h;
      const bf16* kr = kb + (size_t)mt * 16 * KSTR;
      v16bf b0 = join16(*(const v8bf*)(kr),      *(const v8bf*)(kr + 8));
      v16bf b1 = join16(*(const v8bf*)(kr + 32), *(const v8bf*)(kr + 40));
      v8f acc = {};
      acc = wmma_bf16(a0, b0, acc);
      acc = wmma_bf16(a1, b1, acc);
      bool valid = (mt * 16 + mcol) < MM;
      float csum = 0.0f;
      #pragma unroll
      for (int v = 0; v < 8; v++) {
        float p = valid ? __expf(acc[v] * scale - rmax[v]) * rinv[v] : 0.0f;
        csum += p;
        myp[(prow0 + v) * 32 + h * 16 + mcol] = (bf16)p;   // [q_row][m_local]
      }
      atomicAdd(&s_sum[mt * 16 + mcol], csum);             // ds_add_f32
    }
    asm volatile("s_wait_asynccnt 0x0" ::: "memory");
    __syncthreads();                                 // V tile visible, P stores landed
    // P tile as A operand (LDS transpose: D layout is column-per-lane)
    v16bf a2;
    {
      const bf16* pr = myp + (size_t)mcol * 32 + khalf * 16;
      a2 = join16(*(const v8bf*)(pr), *(const v8bf*)(pr + 8));
    }
    #pragma unroll
    for (int dt = 0; dt < 4; dt++) {
      const bf16* vr = s_v + (size_t)(dt * 16 + mcol) * VSTR + khalf * 16;
      v16bf b2 = join16(*(const v8bf*)(vr), *(const v8bf*)(vr + 8));
      oacc[dt] = wmma_bf16(a2, b2, oacc[dt]);
    }
  }

  // ---- epilogue: out (B, Dv, N) and global key-mass flush ----
  float* ob = out + (size_t)b * 64 * NN;
  #pragma unroll
  for (int dt = 0; dt < 4; dt++) {
    float* op = ob + (size_t)(dt * 16 + mcol) * NN + n0 + prow0;
    *(float4*)(op)     = make_float4(oacc[dt][0], oacc[dt][1], oacc[dt][2], oacc[dt][3]);
    *(float4*)(op + 4) = make_float4(oacc[dt][4], oacc[dt][5], oacc[dt][6], oacc[dt][7]);
  }
  __syncthreads();
  for (int i = t; i < MM; i += 512)
    atomicAdd(&ssum_g[b * MPAD + i], s_sum[i]);
}

// ---------------- kernels: scores image ----------------
__global__ void k_zero_scores(float* sc) {
  int i = blockIdx.x * 256 + threadIdx.x;
  if (i < BB * NN) sc[i] = 0.0f;
}
__global__ void k_scatter(const float* __restrict__ ssum_g,
                          const int* __restrict__ idx,
                          float* __restrict__ sc) {
  int i = blockIdx.x * 256 + threadIdx.x;
  if (i < BB * MM) {
    int b = i / MM, m = i % MM;
    sc[b * NN + idx[i]] = ssum_g[b * MPAD + m];
  }
}

// ---------------- host launcher ----------------
extern "C" void kernel_launch(void* const* d_in, const int* in_sizes, int n_in,
                              void* d_out, int out_size, void* d_ws, size_t ws_size,
                              hipStream_t stream) {
  (void)in_sizes; (void)n_in; (void)out_size; (void)ws_size;
  const float* x   = (const float*)d_in[0];
  const float* Wq  = (const float*)d_in[1];
  const float* Wk  = (const float*)d_in[2];
  const float* Wv  = (const float*)d_in[3];
  const int*   idx = (const int*)d_in[4];

  char* ws = (char*)d_ws;
  bf16*  q_bf  = (bf16*)(ws);                       // B*N*64 bf16   = 8,388,608 B
  bf16*  k_bf  = (bf16*)(ws + 8388608);             // B*MPAD*64     =   851,968 B
  bf16*  vt_bf = (bf16*)(ws + 9240576);             // B*64*MPAD     =   851,968 B
  float* ssum  = (float*)(ws + 10092544);           // B*MPAD f32    =    26,624 B

  float* out = (float*)d_out;                       // (B, 64, H, W)
  float* sc  = out + (size_t)BB * 64 * NN;          // (B, 1, H, W)

  k_zero_ssum  <<<(BB * MPAD + 255) / 256, 256, 0, stream>>>(ssum);
  k_proj_q     <<<BB * (NN / 128), 256, 0, stream>>>(x, Wq, q_bf);
  k_proj_kv    <<<BB * (MPAD / 128), 256, 0, stream>>>(x, Wk, Wv, idx, k_bf, vt_bf);
  k_attn       <<<BB * (NN / 256), 512, 0, stream>>>(q_bf, k_bf, vt_bf, out, ssum);
  k_zero_scores<<<(BB * NN + 255) / 256, 256, 0, stream>>>(sc);
  k_scatter    <<<(BB * MM + 255) / 256, 256, 0, stream>>>(ssum, idx, sc);
}
